// MultiHeadAttention_18794776887493
// MI455X (gfx1250) — matrix operand order repaired and checked by reading, not compiled, not run on hardware
//
#include <hip/hip_runtime.h>
#include <hip/hip_bf16.h>
#include <stdint.h>

// Problem constants (B=2, L=2048, D=1024, H=16, HD=64)
#define BATCH 2
#define SEQ   2048
#define DIM   1024
#define NHEAD 16
#define HDIM  64
#define BL    (BATCH*SEQ)   // 4096

typedef __attribute__((ext_vector_type(16))) __bf16 v16bf;
typedef __attribute__((ext_vector_type(8)))  float  v8f;
typedef __attribute__((ext_vector_type(4)))  int    v4i;

// address-space qualified pointer types for the async-to-LDS builtin
typedef v4i __attribute__((address_space(1)))* gas_v4i_p;   // global
typedef v4i __attribute__((address_space(3)))* las_v4i_p;   // LDS

union FragB16 {
    uint4    q[2];
    uint32_t u[8];
    unsigned short s[16];
    v16bf    v;
};

__device__ __forceinline__ unsigned short f2bf(float f) {
    union { float f; uint32_t u; } c; c.f = f;
    uint32_t r = c.u + 0x7fffu + ((c.u >> 16) & 1u);   // round-to-nearest-even
    return (unsigned short)(r >> 16);
}

__device__ __forceinline__ v8f wmma_bf16(const FragB16& a, const FragB16& b, v8f c) {
    return __builtin_amdgcn_wmma_f32_16x16x32_bf16(
        /*neg_a=*/false, a.v, /*neg_b=*/false, b.v,
        /*c_mod=*/(short)0, c, /*reuse_a=*/false, /*reuse_b=*/false);
}

// ---------------------------------------------------------------------------
// CDNA5 async memory->LDS copy (ASYNCcnt-tracked), with safe fallback.
// ---------------------------------------------------------------------------
#if __has_builtin(__builtin_amdgcn_global_load_async_to_lds_b128)
#define USE_ASYNC_LDS 1
#else
#define USE_ASYNC_LDS 0
#endif

__device__ __forceinline__ void cp_b128(unsigned short* l, const unsigned short* g) {
#if USE_ASYNC_LDS
    __builtin_amdgcn_global_load_async_to_lds_b128(
        (gas_v4i_p)(g), (las_v4i_p)(l), /*offset=*/0, /*cpol=*/0);
#else
    *(uint4*)l = *(const uint4*)g;
#endif
}

__device__ __forceinline__ void wait_async_prev() {   // allow 1 batch (4 ops) in flight
#if USE_ASYNC_LDS
#if __has_builtin(__builtin_amdgcn_s_wait_asynccnt)
    __builtin_amdgcn_s_wait_asynccnt(4);
#else
    asm volatile("s_wait_asynccnt 0x4" ::: "memory");
#endif
#endif
}

__device__ __forceinline__ void wait_async_all() {
#if USE_ASYNC_LDS
#if __has_builtin(__builtin_amdgcn_s_wait_asynccnt)
    __builtin_amdgcn_s_wait_asynccnt(0);
#else
    asm volatile("s_wait_asynccnt 0x0" ::: "memory");
#endif
#endif
}

// ---------------------------------------------------------------------------
// f32 -> bf16 conversion (vectorized by 4)
// ---------------------------------------------------------------------------
__global__ __launch_bounds__(256)
void mha_cvt_bf16(const float* __restrict__ src, unsigned short* __restrict__ dst, int n) {
    int i = (blockIdx.x * 256 + threadIdx.x) * 4;
    if (i < n) {
        float4 f = *(const float4*)(src + i);
        ushort4 o;
        o.x = f2bf(f.x); o.y = f2bf(f.y); o.z = f2bf(f.z); o.w = f2bf(f.w);
        *(ushort4*)(dst + i) = o;
    }
}

// ---------------------------------------------------------------------------
// Shared GEMM building blocks. Workgroup tile 128(M) x 128(N), K-step 32,
// 8 waves each owning 32x64 (2x4 WMMA fragments). Double-buffered LDS fed
// by async global->LDS copies.
// ---------------------------------------------------------------------------
#define A_TILE (128 * 32)
#define B_TILE (128 * 32)

__device__ __forceinline__ void gemm_issue(const unsigned short* __restrict__ A,
                                           const unsigned short* __restrict__ W,
                                           unsigned short* aL, unsigned short* bL,
                                           int m0, int n0, int k0, int tid) {
    const int r0 = tid >> 2, c0 = (tid & 3) * 8;
    cp_b128(&aL[r0 * 32 + c0],        &A[(size_t)(m0 + r0) * DIM + k0 + c0]);
    cp_b128(&aL[(r0 + 64) * 32 + c0], &A[(size_t)(m0 + 64 + r0) * DIM + k0 + c0]);
    cp_b128(&bL[r0 * 32 + c0],        &W[(size_t)(n0 + r0) * DIM + k0 + c0]);
    cp_b128(&bL[(r0 + 64) * 32 + c0], &W[(size_t)(n0 + 64 + r0) * DIM + k0 + c0]);
}

__device__ __forceinline__ void gemm_compute(const unsigned short* aL,
                                             const unsigned short* bL,
                                             int mw, int nw, int l15, int half,
                                             v8f acc[2][4]) {
    FragB16 af[2], bf[4];
#pragma unroll
    for (int mt = 0; mt < 2; ++mt) {
        const unsigned short* p = &aL[(mw + mt * 16 + l15) * 32 + half * 8];
        af[mt].q[0] = *(const uint4*)p;          // k = base .. base+7
        af[mt].q[1] = *(const uint4*)(p + 16);   // k = base+16 .. base+23
    }
#pragma unroll
    for (int nt = 0; nt < 4; ++nt) {
        const unsigned short* p = &bL[(nw + nt * 16 + l15) * 32 + half * 8];
        bf[nt].q[0] = *(const uint4*)p;          // k = base .. base+7
        bf[nt].q[1] = *(const uint4*)(p + 16);   // k = base+16 .. base+23
    }
#pragma unroll
    for (int mt = 0; mt < 2; ++mt)
#pragma unroll
        for (int nt = 0; nt < 4; ++nt)
            acc[mt][nt] = wmma_bf16(af[mt], bf[nt], acc[mt][nt]);
}

// ---------------------------------------------------------------------------
// GEMM1: qkv[m,n] = sum_k xb[m,k] * wqkv[n,k]   (M=4096, N=3072, K=1024)
// Epilogue scatters bf16 into Q [b,h,l,64], K [b,h,l,64], Vt [b,h,64,l].
// ---------------------------------------------------------------------------
__global__ __launch_bounds__(256)
void mha_gemm_qkv(const unsigned short* __restrict__ A,   // [4096,1024] bf16
                  const unsigned short* __restrict__ W,   // [3072,1024] bf16
                  unsigned short* __restrict__ qbuf,
                  unsigned short* __restrict__ kbuf,
                  unsigned short* __restrict__ vt) {
    __shared__ __align__(16) unsigned short aLds[2 * A_TILE];
    __shared__ __align__(16) unsigned short bLds[2 * B_TILE];

    const int tid  = threadIdx.x;
    const int wave = tid >> 5;
    const int lane = tid & 31;
    const int half = lane >> 4;
    const int l15  = lane & 15;
    const int m0 = (blockIdx.x & 31) * 128;     // 32 M-blocks
    const int n0 = (blockIdx.x >> 5) * 128;     // 24 N-blocks
    const int mw = (wave & 3) * 32;
    const int nw = (wave >> 2) * 64;

    v8f acc[2][4];
#pragma unroll
    for (int i = 0; i < 2; ++i)
#pragma unroll
        for (int j = 0; j < 4; ++j)
#pragma unroll
            for (int r = 0; r < 8; ++r) acc[i][j][r] = 0.0f;

    gemm_issue(A, W, aLds, bLds, m0, n0, 0, tid);
    for (int it = 0; it < DIM / 32; ++it) {
        const int cur = it & 1;
        if (it + 1 < DIM / 32) {
            gemm_issue(A, W, aLds + (cur ^ 1) * A_TILE, bLds + (cur ^ 1) * B_TILE,
                       m0, n0, (it + 1) * 32, tid);
            wait_async_prev();
        } else {
            wait_async_all();
        }
        __syncthreads();
        gemm_compute(aLds + cur * A_TILE, bLds + cur * B_TILE, mw, nw, l15, half, acc);
        __syncthreads();
    }

    // scatter epilogue: split into Q / K / V-transposed, bf16
#pragma unroll
    for (int mt = 0; mt < 2; ++mt) {
#pragma unroll
        for (int nt = 0; nt < 4; ++nt) {
            int ncol  = n0 + nw + nt * 16 + l15;
            int three = ncol >> 10;
            int h     = (ncol >> 6) & 15;
            int hd    = ncol & 63;
#pragma unroll
            for (int r = 0; r < 8; ++r) {
                int mrow = m0 + mw + mt * 16 + r + half * 8;
                int b = mrow >> 11, l = mrow & (SEQ - 1);
                unsigned short bv = f2bf(acc[mt][nt][r]);
                size_t bh = (size_t)(b * NHEAD + h);
                if (three == 0)      qbuf[(bh * SEQ + l) * HDIM + hd] = bv;
                else if (three == 1) kbuf[(bh * SEQ + l) * HDIM + hd] = bv;
                else                 vt[(bh * HDIM + hd) * SEQ + l]   = bv;
            }
        }
    }
}

// ---------------------------------------------------------------------------
// Flash attention (causal). Grid (32 bh, 32 qblocks), 128 threads = 4 waves,
// each wave owns a 16-query tile, streams 32-key chunks with online softmax.
// ---------------------------------------------------------------------------
__global__ __launch_bounds__(128)
void mha_attn(const unsigned short* __restrict__ qbuf,
              const unsigned short* __restrict__ kbuf,
              const unsigned short* __restrict__ vt,
              unsigned short* __restrict__ aout) {   // [4096, 1024] bf16
    __shared__ __align__(16) unsigned short pLds[4][16 * 32];

    const int tid  = threadIdx.x;
    const int wave = tid >> 5;
    const int lane = tid & 31;
    const int half = lane >> 4;
    const int l15  = lane & 15;
    const int bh   = blockIdx.x;               // b*16 + h
    const int b    = bh >> 4, h = bh & 15;
    const int q0   = blockIdx.y * 64 + wave * 16;

    const unsigned short* Q = qbuf + (size_t)bh * SEQ * HDIM;
    const unsigned short* K = kbuf + (size_t)bh * SEQ * HDIM;
    const unsigned short* V = vt   + (size_t)bh * HDIM * SEQ;
    unsigned short* P = pLds[wave];

    // Q A-fragments for both 32-wide k-steps across hd=0..63
    FragB16 qf[2];
#pragma unroll
    for (int s = 0; s < 2; ++s) {
        const unsigned short* p = &Q[(size_t)(q0 + l15) * HDIM + s * 32 + half * 8];
        qf[s].q[0] = *(const uint4*)p;
        qf[s].q[1] = *(const uint4*)(p + 16);
    }

    v8f oacc[4];
#pragma unroll
    for (int nt = 0; nt < 4; ++nt)
#pragma unroll
        for (int r = 0; r < 8; ++r) oacc[nt][r] = 0.0f;
    float m_i[8], l_i[8];
#pragma unroll
    for (int r = 0; r < 8; ++r) { m_i[r] = -3.0e38f; l_i[r] = 0.0f; }

    const int nchunks = (q0 + 47) >> 5;        // ceil((q0+16)/32)
    for (int c = 0; c < nchunks; ++c) {
        const int k0 = c * 32;
        // ---- scores S = Q K^T (16 x 32 tile, two 16-wide n-tiles) ----
        v8f sc[2];
#pragma unroll
        for (int nt = 0; nt < 2; ++nt)
#pragma unroll
            for (int r = 0; r < 8; ++r) sc[nt][r] = 0.0f;
#pragma unroll
        for (int nt = 0; nt < 2; ++nt) {
#pragma unroll
            for (int s = 0; s < 2; ++s) {
                FragB16 kf;
                const unsigned short* p =
                    &K[(size_t)(k0 + nt * 16 + l15) * HDIM + s * 32 + half * 8];
                kf.q[0] = *(const uint4*)p;
                kf.q[1] = *(const uint4*)(p + 16);
                sc[nt] = wmma_bf16(qf[s], kf, sc[nt]);
            }
        }
        // ---- scale + causal mask (only diagonal chunk needs the mask) ----
        const bool maskchunk = (k0 + 31 > q0);
#pragma unroll
        for (int nt = 0; nt < 2; ++nt)
#pragma unroll
            for (int r = 0; r < 8; ++r) {
                float s = sc[nt][r] * 0.125f;            // 1/sqrt(64)
                if (maskchunk) {
                    int row = q0 + r + half * 8;
                    int col = k0 + nt * 16 + l15;
                    if (col > row) s = -3.0e38f;
                }
                sc[nt][r] = s;
            }
        // ---- online softmax (row reductions across 16-lane column group) ----
#pragma unroll
        for (int r = 0; r < 8; ++r) {
            float mx = fmaxf(sc[0][r], sc[1][r]);
            mx = fmaxf(mx, __shfl_xor(mx, 1, 32));
            mx = fmaxf(mx, __shfl_xor(mx, 2, 32));
            mx = fmaxf(mx, __shfl_xor(mx, 4, 32));
            mx = fmaxf(mx, __shfl_xor(mx, 8, 32));
            float nm   = fmaxf(m_i[r], mx);
            float corr = __expf(m_i[r] - nm);
            m_i[r] = nm;
            float p0 = __expf(sc[0][r] - nm);
            float p1 = __expf(sc[1][r] - nm);
            sc[0][r] = p0; sc[1][r] = p1;
            float rs = p0 + p1;
            rs += __shfl_xor(rs, 1, 32);
            rs += __shfl_xor(rs, 2, 32);
            rs += __shfl_xor(rs, 4, 32);
            rs += __shfl_xor(rs, 8, 32);
            l_i[r] = l_i[r] * corr + rs;
#pragma unroll
            for (int nt = 0; nt < 4; ++nt) oacc[nt][r] *= corr;
        }
        // ---- P: C-fragment layout -> A-fragment layout via per-wave LDS ----
#pragma unroll
        for (int nt = 0; nt < 2; ++nt)
#pragma unroll
            for (int r = 0; r < 8; ++r)
                P[(r + half * 8) * 32 + nt * 16 + l15] = f2bf(sc[nt][r]);
        asm volatile("s_wait_dscnt 0x0" ::: "memory");
        FragB16 pf;
        {
            const unsigned short* p = &P[l15 * 32 + half * 8];
            pf.q[0] = *(const uint4*)p;
            pf.q[1] = *(const uint4*)(p + 16);
        }
        // ---- O += P V  (Vt gives contiguous per-lane B-operand reads) ----
#pragma unroll
        for (int nt = 0; nt < 4; ++nt) {
            FragB16 vf;
            const unsigned short* p =
                &V[(size_t)(nt * 16 + l15) * SEQ + k0 + half * 8];
            vf.q[0] = *(const uint4*)p;
            vf.q[1] = *(const uint4*)(p + 16);
            oacc[nt] = wmma_bf16(pf, vf, oacc[nt]);
        }
    }

    // ---- normalize + store bf16 [b*2048+l, h*64+hd] ----
#pragma unroll
    for (int r = 0; r < 8; ++r) {
        float inv = 1.0f / l_i[r];
#pragma unroll
        for (int nt = 0; nt < 4; ++nt) oacc[nt][r] *= inv;
    }
#pragma unroll
    for (int nt = 0; nt < 4; ++nt)
#pragma unroll
        for (int r = 0; r < 8; ++r) {
            int l = q0 + r + half * 8;
            aout[((size_t)(b * SEQ + l)) * DIM + h * HDIM + nt * 16 + l15] =
                f2bf(oacc[nt][r]);
        }
}

// ---------------------------------------------------------------------------
// GEMM2: out[m,n] = sum_k ao[m,k] * wout[n,k]   (M=4096, N=1024, K=1024), f32 out
// ---------------------------------------------------------------------------
__global__ __launch_bounds__(256)
void mha_gemm_out(const unsigned short* __restrict__ A,   // [4096,1024] bf16
                  const unsigned short* __restrict__ W,   // [1024,1024] bf16
                  float* __restrict__ out) {
    __shared__ __align__(16) unsigned short aLds[2 * A_TILE];
    __shared__ __align__(16) unsigned short bLds[2 * B_TILE];

    const int tid  = threadIdx.x;
    const int wave = tid >> 5;
    const int lane = tid & 31;
    const int half = lane >> 4;
    const int l15  = lane & 15;
    const int m0 = (blockIdx.x & 31) * 128;
    const int n0 = (blockIdx.x >> 5) * 128;     // 8 N-blocks
    const int mw = (wave & 3) * 32;
    const int nw = (wave >> 2) * 64;

    v8f acc[2][4];
#pragma unroll
    for (int i = 0; i < 2; ++i)
#pragma unroll
        for (int j = 0; j < 4; ++j)
#pragma unroll
            for (int r = 0; r < 8; ++r) acc[i][j][r] = 0.0f;

    gemm_issue(A, W, aLds, bLds, m0, n0, 0, tid);
    for (int it = 0; it < DIM / 32; ++it) {
        const int cur = it & 1;
        if (it + 1 < DIM / 32) {
            gemm_issue(A, W, aLds + (cur ^ 1) * A_TILE, bLds + (cur ^ 1) * B_TILE,
                       m0, n0, (it + 1) * 32, tid);
            wait_async_prev();
        } else {
            wait_async_all();
        }
        __syncthreads();
        gemm_compute(aLds + cur * A_TILE, bLds + cur * B_TILE, mw, nw, l15, half, acc);
        __syncthreads();
    }

#pragma unroll
    for (int mt = 0; mt < 2; ++mt)
#pragma unroll
        for (int nt = 0; nt < 4; ++nt) {
            int ncol = n0 + nw + nt * 16 + l15;
#pragma unroll
            for (int r = 0; r < 8; ++r) {
                int mrow = m0 + mw + mt * 16 + r + half * 8;
                out[(size_t)mrow * DIM + ncol] = acc[mt][nt][r];
            }
        }
}

// ---------------------------------------------------------------------------
extern "C" void kernel_launch(void* const* d_in, const int* in_sizes, int n_in,
                              void* d_out, int out_size, void* d_ws, size_t ws_size,
                              hipStream_t stream) {
    const float* x     = (const float*)d_in[0];   // [2,2048,1024]
    const float* w_qkv = (const float*)d_in[1];   // [3072,1024]
    const float* w_out = (const float*)d_in[2];   // [1024,1024]
    float* out = (float*)d_out;                   // [2,2048,1024]

    // workspace layout (ushort/bf16 elements)
    unsigned short* ws  = (unsigned short*)d_ws;
    unsigned short* xb  = ws;                                  // 4096*1024
    unsigned short* wqb = xb  + (size_t)BL * DIM;              // 3072*1024
    unsigned short* wob = wqb + (size_t)3 * DIM * DIM;         // 1024*1024
    unsigned short* qb  = wob + (size_t)DIM * DIM;             // 2*16*2048*64
    unsigned short* kb  = qb  + (size_t)BATCH * NHEAD * SEQ * HDIM;
    unsigned short* vtb = kb  + (size_t)BATCH * NHEAD * SEQ * HDIM;
    unsigned short* ao  = vtb + (size_t)BATCH * NHEAD * SEQ * HDIM;  // 4096*1024

    // 1) f32 -> bf16 conversions
    mha_cvt_bf16<<<(BL * DIM) / 1024, 256, 0, stream>>>(x, xb, BL * DIM);
    mha_cvt_bf16<<<(3 * DIM * DIM) / 1024, 256, 0, stream>>>(w_qkv, wqb, 3 * DIM * DIM);
    mha_cvt_bf16<<<(DIM * DIM) / 1024, 256, 0, stream>>>(w_out, wob, DIM * DIM);

    // 2) QKV projection (M=4096, N=3072): 32 x 24 = 768 workgroups
    mha_gemm_qkv<<<dim3(32 * 24), 256, 0, stream>>>(xb, wqb, qb, kb, vtb);

    // 3) causal flash attention: (B*H=32) x (L/64=32) workgroups, 4 waves
    mha_attn<<<dim3(32, 32), 128, 0, stream>>>(qb, kb, vtb, ao);

    // 4) output projection (M=4096, N=1024): 32 x 8 = 256 workgroups
    mha_gemm_out<<<dim3(32 * 8), 256, 0, stream>>>(ao, wob, out);
}
